// GraphPropMatchingLayer_85633057948082
// MI455X (gfx1250) — matrix operand order repaired and checked by reading, not compiled, not run on hardware
//
#include <hip/hip_runtime.h>
#include <hip/hip_bf16.h>
#include <math.h>

// ---------------------------------------------------------------------------
// Problem constants (from reference)
// ---------------------------------------------------------------------------
#define N_NODES   32768
#define N_PER     2048          // nodes per graph
#define N_PAIRS   8             // graph pairs for cross attention
#define N_EDGES   262144
#define DIM       128           // node feature dim
#define DE        64            // edge feature dim
#define HH        256           // hidden dim (edge + node MLPs)
#define KEDGE     320           // 2*DIM + DE
#define KNODE     512           // HH + DIM + DIM

#define EPB       64            // edges per block (edge MLP M-tile)
#define NPB       32            // nodes per block (node MLP M-tile)

typedef __attribute__((ext_vector_type(16))) __bf16 v16bf;
typedef __attribute__((ext_vector_type(8)))  float  v8f;

__device__ __forceinline__ v8f wmma_bf16(v16bf a, v16bf b, v8f c) {
  // D = A(16x32) * B(32x16) + C, f32 accumulate
  return __builtin_amdgcn_wmma_f32_16x16x32_bf16(
      /*neg_a=*/false, a, /*neg_b=*/false, b,
      /*c_mod=*/(short)0, c, /*reuse_a=*/false, /*reuse_b=*/false);
}

// ---------------------------------------------------------------------------
// WMMA operand loaders (contiguous k order, same in A and B, wave32)
// A 16x32 bf16 : M = lane&15 ; elem t -> K = t + 16*(lane>=16)
//   i.e. contiguous run [16h, 16h+16)
// B 32x16 bf16 : N = lane&15 ; elem t -> K = t + 16*(lane>=16)
// C 16x16 f32  : value r    -> M = r + 8*(lane>=16), N = lane&15
// ---------------------------------------------------------------------------

// A tile from bf16 buffer (LDS or global), row-major [..][stride]
__device__ __forceinline__ v16bf load_A_bf(const __bf16* s, int stride,
                                           int mbase, int kbase, int lane) {
  const int h = (lane & 16) ? 16 : 0;
  const __bf16* p = s + (size_t)(mbase + (lane & 15)) * stride + kbase + h;
  v16bf a;
#pragma unroll
  for (int t = 0; t < 8; ++t) a[t] = p[t];
#pragma unroll
  for (int t = 0; t < 8; ++t) a[8 + t] = p[8 + t];
  return a;
}

// B tile where B[k][n] = Wrow_n[kbase+k]  (B = W^T, W row-major (N,K), bf16)
__device__ __forceinline__ v16bf load_B_rowsT_bf(const __bf16* W, int nrow_base,
                                                 int kbase, int lane, int ld) {
  const __bf16* p = W + (size_t)(nrow_base + (lane & 15)) * ld + kbase +
                    ((lane & 16) ? 16 : 0);
  v16bf b;
#pragma unroll
  for (int t = 0; t < 16; ++t) b[t] = p[t];
  return b;
}

// B tile where B[k][n] = Y[jbase+k][dbase+n]  (strided column reads, bf16)
__device__ __forceinline__ v16bf load_B_cols_bf(const __bf16* Y, int jbase,
                                                int dbase, int lane, int ld) {
  const int n  = lane & 15;
  const int kb = (lane & 16) ? 16 : 0;
  v16bf b;
#pragma unroll
  for (int t = 0; t < 16; ++t)
    b[t] = Y[(size_t)(jbase + kb + t) * ld + dbase + n];
  return b;
}

// ---------------------------------------------------------------------------
// Workspace init / precompute
// ---------------------------------------------------------------------------
__global__ void zero_kernel(float* __restrict__ p, int n) {
  for (int i = blockIdx.x * blockDim.x + threadIdx.x; i < n;
       i += gridDim.x * blockDim.x)
    p[i] = 0.0f;
}

__global__ void cvt_bf16_kernel(const float* __restrict__ src,
                                __bf16* __restrict__ dst, int n) {
  for (int i = blockIdx.x * blockDim.x + threadIdx.x; i < n;
       i += gridDim.x * blockDim.x)
    dst[i] = (__bf16)src[i];
}

// ---------------------------------------------------------------------------
// Fused edge MLP + scatter-add (all operands bf16).
//   in = concat(nbf[gA[e]], nbf[gB[e]], efbf[e])      (320)
//   hid = relu(in @ W1^T + b1)                         (256)
//   out = hid @ W2^T + b2                              (256)
//   agg[sc[e]] += out  (atomic f32)
// Block: 512 threads = 16 waves; 64 edges/block.
// Wave w owns out cols [16w,16w+16) for all 4 M-subtiles -> every B tile is
// reused 4x; weight slices stream from L2 1/4 as often as with 16 edges/blk.
// ---------------------------------------------------------------------------
__global__ void __launch_bounds__(512)
edge_mlp_kernel(const __bf16* __restrict__ nbf, const __bf16* __restrict__ efbf,
                const int* __restrict__ gA, const int* __restrict__ gB,
                const int* __restrict__ sc,
                const __bf16* __restrict__ W1, const float* __restrict__ b1,
                const __bf16* __restrict__ W2, const float* __restrict__ b2,
                float* __restrict__ agg) {
  __shared__ __bf16 sIn[EPB * KEDGE];   // 40 KB
  __shared__ __bf16 sHid[EPB * HH];     // 32 KB

  const int eb = blockIdx.x * EPB;

  // Stage 0: gather + concat into LDS, 4 bf16 (uint2) per copy.
  // Row = 80 uint2: [0,32) from nbf[gA], [32,64) from nbf[gB], [64,80) from ef.
  for (int idx = threadIdx.x; idx < EPB * 80; idx += 512) {
    const int m = idx / 80;
    const int j = idx - m * 80;
    const int e = eb + m;
    uint2 v;
    if (j < 32)      v = ((const uint2*)(nbf + (size_t)gA[e] * DIM))[j];
    else if (j < 64) v = ((const uint2*)(nbf + (size_t)gB[e] * DIM))[j - 32];
    else             v = ((const uint2*)(efbf + (size_t)e * DE))[j - 64];
    ((uint2*)sIn)[idx] = v;
  }
  __syncthreads();

  const int wave  = threadIdx.x >> 5;
  const int lane  = threadIdx.x & 31;
  const int half8 = (lane & 16) ? 8 : 0;
  const int N     = lane & 15;
  const int nc    = wave * 16;

  // Stage 1: hidden = relu(in @ W1^T + b1)
  {
    v8f c[EPB / 16] = {};
#pragma unroll
    for (int kt = 0; kt < KEDGE / 32; ++kt) {
      const v16bf b = load_B_rowsT_bf(W1, nc, kt * 32, lane, KEDGE);
#pragma unroll
      for (int mt = 0; mt < EPB / 16; ++mt)
        c[mt] = wmma_bf16(load_A_bf(sIn, KEDGE, mt * 16, kt * 32, lane), b,
                          c[mt]);
    }
    const float bias = b1[nc + N];
#pragma unroll
    for (int mt = 0; mt < EPB / 16; ++mt)
#pragma unroll
      for (int r = 0; r < 8; ++r)
        sHid[(mt * 16 + r + half8) * HH + nc + N] =
            (__bf16)fmaxf(c[mt][r] + bias, 0.0f);
  }
  __syncthreads();

  // Stage 2: out = hid @ W2^T + b2 ; atomic scatter into agg
  {
    v8f c[EPB / 16] = {};
#pragma unroll
    for (int kt = 0; kt < HH / 32; ++kt) {
      const v16bf b = load_B_rowsT_bf(W2, nc, kt * 32, lane, HH);
#pragma unroll
      for (int mt = 0; mt < EPB / 16; ++mt)
        c[mt] = wmma_bf16(load_A_bf(sHid, HH, mt * 16, kt * 32, lane), b,
                          c[mt]);
    }
    const float bias = b2[nc + N];
#pragma unroll
    for (int mt = 0; mt < EPB / 16; ++mt)
#pragma unroll
      for (int r = 0; r < 8; ++r) {
        const int e = eb + mt * 16 + r + half8;
        atomicAdd(&agg[(size_t)sc[e] * HH + nc + N], c[mt][r] + bias);
      }
  }
}

// ---------------------------------------------------------------------------
// Attention softmax stats (online max / sum-exp per row), bf16 operands.
//   swap==0: rows = X (graph 2p), cols = Y (graph 2p+1)  -> row softmax stats
//   swap==1: rows = Y, cols = X                           -> col softmax stats
// Block: 256 threads = 8 waves; each wave owns one (pair,row-tile) task.
// ---------------------------------------------------------------------------
__global__ void __launch_bounds__(256)
attn_stats_kernel(const __bf16* __restrict__ nbf, float* __restrict__ out_m,
                  float* __restrict__ out_l, int swap) {
  const int wave = threadIdx.x >> 5;
  const int lane = threadIdx.x & 31;
  const int task = blockIdx.x * 8 + wave;    // 0..1023
  const int pair = task >> 7;
  const int rt   = task & 127;

  const __bf16* X = nbf + (size_t)(pair * 2 * N_PER + (swap ? N_PER : 0)) * DIM;
  const __bf16* Y = nbf + (size_t)(pair * 2 * N_PER + (swap ? 0 : N_PER)) * DIM;

  const int half8 = (lane & 16) ? 8 : 0;

  v16bf a[4];
#pragma unroll
  for (int kt = 0; kt < 4; ++kt)
    a[kt] = load_A_bf(X, DIM, rt * 16, kt * 32, lane);

  float mr[8], lr[8];
#pragma unroll
  for (int r = 0; r < 8; ++r) { mr[r] = -INFINITY; lr[r] = 0.0f; }

  for (int jt = 0; jt < N_PER / 16; ++jt) {
    v8f c = {};
#pragma unroll
    for (int kt = 0; kt < 4; ++kt)
      c = wmma_bf16(a[kt], load_B_rowsT_bf(Y, jt * 16, kt * 32, lane, DIM), c);
#pragma unroll
    for (int r = 0; r < 8; ++r) {
      const float s = c[r];
      if (s > mr[r]) { lr[r] = lr[r] * __expf(mr[r] - s) + 1.0f; mr[r] = s; }
      else           { lr[r] += __expf(s - mr[r]); }
    }
  }

  // reduce across the 16 lanes that share each row (stays within half-wave)
#pragma unroll
  for (int off = 1; off < 16; off <<= 1) {
#pragma unroll
    for (int r = 0; r < 8; ++r) {
      const float mo = __shfl_xor(mr[r], off, 32);
      const float lo = __shfl_xor(lr[r], off, 32);
      const float mn = fmaxf(mr[r], mo);
      lr[r] = lr[r] * __expf(mr[r] - mn) + lo * __expf(mo - mn);
      mr[r] = mn;
    }
  }

  if ((lane & 15) == 0) {
#pragma unroll
    for (int r = 0; r < 8; ++r) {
      const int row = pair * N_PER + rt * 16 + r + half8;
      out_m[row] = mr[r];
      out_l[row] = lr[r];
    }
  }
}

// ---------------------------------------------------------------------------
// Attention apply: attin[node] = nodes[node] - (1/l_i) * sum_j exp(s_ij-m_i)*Y[j]
// Block: 256 threads = 8 waves; tasks = pair(8) x rowtile(128) x dtile(8) = 8192.
// ---------------------------------------------------------------------------
__global__ void __launch_bounds__(256)
attn_apply_kernel(const __bf16* __restrict__ nbf, const float* __restrict__ nodes,
                  const float* __restrict__ st_m, const float* __restrict__ st_l,
                  float* __restrict__ attin, int swap) {
  __shared__ __bf16 sP[8][16 * 32];   // per-wave C->A layout staging (1KB each)

  const int wave = threadIdx.x >> 5;
  const int lane = threadIdx.x & 31;
  const int task = blockIdx.x * 8 + wave;  // 0..8191
  const int pair = task >> 10;
  const int rem  = task & 1023;
  const int rt   = rem >> 3;
  const int dt   = rem & 7;

  const int xoff = pair * 2 * N_PER + (swap ? N_PER : 0);
  const __bf16* X = nbf + (size_t)xoff * DIM;
  const __bf16* Y = nbf + (size_t)(pair * 2 * N_PER + (swap ? 0 : N_PER)) * DIM;

  const int half8 = (lane & 16) ? 8 : 0;
  const int N     = lane & 15;

  float rm[8], rl[8];
#pragma unroll
  for (int r = 0; r < 8; ++r) {
    const int row = pair * N_PER + rt * 16 + r + half8;
    rm[r] = st_m[row];
    rl[r] = st_l[row];
  }

  v16bf a[4];
#pragma unroll
  for (int kt = 0; kt < 4; ++kt)
    a[kt] = load_A_bf(X, DIM, rt * 16, kt * 32, lane);

  __bf16* P = &sP[wave][0];
  v8f acc = {};

  for (int jj = 0; jj < N_PER / 32; ++jj) {
    // two 16-wide score sub-tiles -> one 16x32 P tile (A operand K=32)
#pragma unroll
    for (int sub = 0; sub < 2; ++sub) {
      const int jt = jj * 2 + sub;
      v8f c = {};
#pragma unroll
      for (int kt = 0; kt < 4; ++kt)
        c = wmma_bf16(a[kt], load_B_rowsT_bf(Y, jt * 16, kt * 32, lane, DIM), c);
#pragma unroll
      for (int r = 0; r < 8; ++r)
        P[(r + half8) * 32 + sub * 16 + N] = (__bf16)__expf(c[r] - rm[r]);
    }
    // same-wave LDS ops are in-order; compiler inserts s_wait_dscnt for the
    // register dependency of the A-operand load below.
    acc = wmma_bf16(load_A_bf(P, 32, 0, 0, lane),
                    load_B_cols_bf(Y, jj * 32, dt * 16, lane, DIM), acc);
  }

#pragma unroll
  for (int r = 0; r < 8; ++r) {
    const int node = xoff + rt * 16 + r + half8;
    const float v  = acc[r] / rl[r];
    attin[(size_t)node * DIM + dt * 16 + N] =
        nodes[(size_t)node * DIM + dt * 16 + N] - v;
  }
}

// ---------------------------------------------------------------------------
// Fused node MLP with residual:
//   in  = concat(agg[n](256), attin[n](128), nodes[n](128))   (512)
//   hid = relu(in @ Wn1^T + bn1)                               (256)
//   out = nodes[n] + hid @ Wn2^T + bn2                         (128)
// Block: 512 threads = 16 waves; 32 nodes/block (2 M-subtiles, B reused 2x).
// Stage 2 uses all 16 waves: wave = (mtile = w>>3, coltile = w&7).
// ---------------------------------------------------------------------------
__global__ void __launch_bounds__(512)
node_mlp_kernel(const float* __restrict__ nodes, const __bf16* __restrict__ nbf,
                const float* __restrict__ agg, const float* __restrict__ attin,
                const __bf16* __restrict__ Wn1, const float* __restrict__ bn1,
                const __bf16* __restrict__ Wn2, const float* __restrict__ bn2,
                float* __restrict__ out) {
  __shared__ __bf16 sIn[NPB * KNODE];   // 32 KB
  __shared__ __bf16 sHid[NPB * HH];     // 16 KB

  const int nb = blockIdx.x * NPB;

  for (int idx = threadIdx.x; idx < NPB * KNODE; idx += 512) {
    const int m = idx >> 9;
    const int k = idx & 511;
    const int node = nb + m;
    __bf16 v;
    if (k < HH)            v = (__bf16)agg[(size_t)node * HH + k];
    else if (k < HH + DIM) v = (__bf16)attin[(size_t)node * DIM + (k - HH)];
    else                   v = nbf[(size_t)node * DIM + (k - HH - DIM)];
    sIn[idx] = v;
  }
  __syncthreads();

  const int wave  = threadIdx.x >> 5;
  const int lane  = threadIdx.x & 31;
  const int half8 = (lane & 16) ? 8 : 0;
  const int N     = lane & 15;

  {
    const int nc = wave * 16;
    v8f c[NPB / 16] = {};
#pragma unroll
    for (int kt = 0; kt < KNODE / 32; ++kt) {
      const v16bf b = load_B_rowsT_bf(Wn1, nc, kt * 32, lane, KNODE);
#pragma unroll
      for (int mt = 0; mt < NPB / 16; ++mt)
        c[mt] = wmma_bf16(load_A_bf(sIn, KNODE, mt * 16, kt * 32, lane), b,
                          c[mt]);
    }
    const float bias = bn1[nc + N];
#pragma unroll
    for (int mt = 0; mt < NPB / 16; ++mt)
#pragma unroll
      for (int r = 0; r < 8; ++r)
        sHid[(mt * 16 + r + half8) * HH + nc + N] =
            (__bf16)fmaxf(c[mt][r] + bias, 0.0f);
  }
  __syncthreads();

  {
    const int mt = wave >> 3;           // 0..1
    const int dc = (wave & 7) * 16;     // 0..112
    v8f c = {};
#pragma unroll
    for (int kt = 0; kt < HH / 32; ++kt)
      c = wmma_bf16(load_A_bf(sHid, HH, mt * 16, kt * 32, lane),
                    load_B_rowsT_bf(Wn2, dc, kt * 32, lane, HH), c);
    const float bias = bn2[dc + N];
#pragma unroll
    for (int r = 0; r < 8; ++r) {
      const int node = nb + mt * 16 + r + half8;
      out[(size_t)node * DIM + dc + N] =
          nodes[(size_t)node * DIM + dc + N] + c[r] + bias;
    }
  }
}

// ---------------------------------------------------------------------------
// Launcher
// ---------------------------------------------------------------------------
extern "C" void kernel_launch(void* const* d_in, const int* in_sizes, int n_in,
                              void* d_out, int out_size, void* d_ws,
                              size_t ws_size, hipStream_t stream) {
  const float* nodes = (const float*)d_in[0];
  const float* ef    = (const float*)d_in[1];
  const int*   from  = (const int*)d_in[2];
  const int*   to    = (const int*)d_in[3];
  // d_in[4] = graph_idx (unused: layout is contiguous), d_in[5] = n_graphs
  const float* Wm1 = (const float*)d_in[6];
  const float* bm1 = (const float*)d_in[7];
  const float* Wm2 = (const float*)d_in[8];
  const float* bm2 = (const float*)d_in[9];
  const float* Wr1 = (const float*)d_in[10];
  const float* br1 = (const float*)d_in[11];
  const float* Wr2 = (const float*)d_in[12];
  const float* br2 = (const float*)d_in[13];
  const float* Wn1 = (const float*)d_in[14];
  const float* bn1 = (const float*)d_in[15];
  const float* Wn2 = (const float*)d_in[16];
  const float* bn2 = (const float*)d_in[17];
  float* out = (float*)d_out;

  // Workspace layout: f32 region first, then bf16 region (all 16B aligned).
  float* agg   = (float*)d_ws;                           // 32768*256 f32
  float* attin = agg + (size_t)N_NODES * HH;             // 32768*128 f32
  float* rm    = attin + (size_t)N_NODES * DIM;          // 8*2048 f32 x4
  float* rl    = rm + N_PAIRS * N_PER;
  float* cm    = rl + N_PAIRS * N_PER;
  float* cl    = cm + N_PAIRS * N_PER;
  __bf16* nbf   = (__bf16*)(cl + N_PAIRS * N_PER);       // 32768*128
  __bf16* efbf  = nbf + (size_t)N_NODES * DIM;           // 262144*64
  __bf16* wm1b  = efbf + (size_t)N_EDGES * DE;           // 256*320
  __bf16* wm2b  = wm1b + HH * KEDGE;                     // 256*256
  __bf16* wr1b  = wm2b + HH * HH;                        // 256*320
  __bf16* wr2b  = wr1b + HH * KEDGE;                     // 256*256
  __bf16* wn1b  = wr2b + HH * HH;                        // 256*512
  __bf16* wn2b  = wn1b + HH * KNODE;                     // 128*256

  // 0) precompute bf16 copies (weights are L2-resident afterwards)
  cvt_bf16_kernel<<<2048, 256, 0, stream>>>(nodes, nbf, N_NODES * DIM);
  cvt_bf16_kernel<<<4096, 256, 0, stream>>>(ef, efbf, N_EDGES * DE);
  cvt_bf16_kernel<<<80, 256, 0, stream>>>(Wm1, wm1b, HH * KEDGE);
  cvt_bf16_kernel<<<64, 256, 0, stream>>>(Wm2, wm2b, HH * HH);
  cvt_bf16_kernel<<<80, 256, 0, stream>>>(Wr1, wr1b, HH * KEDGE);
  cvt_bf16_kernel<<<64, 256, 0, stream>>>(Wr2, wr2b, HH * HH);
  cvt_bf16_kernel<<<128, 256, 0, stream>>>(Wn1, wn1b, HH * KNODE);
  cvt_bf16_kernel<<<32, 256, 0, stream>>>(Wn2, wn2b, DIM * HH);

  // 1) zero the aggregation buffer (atomically accumulated)
  zero_kernel<<<2048, 256, 0, stream>>>(agg, N_NODES * HH);

  // 2) edge MLPs with fused scatter-add
  //    fwd: in = [nodes[from], nodes[to], ef], scatter by to
  edge_mlp_kernel<<<N_EDGES / EPB, 512, 0, stream>>>(
      nbf, efbf, from, to, to, wm1b, bm1, wm2b, bm2, agg);
  //    rev: in = [nodes[to], nodes[from], ef], scatter by from
  edge_mlp_kernel<<<N_EDGES / EPB, 512, 0, stream>>>(
      nbf, efbf, to, from, from, wr1b, br1, wr2b, br2, agg);

  // 3) cross-graph attention softmax stats (row and column)
  attn_stats_kernel<<<128, 256, 0, stream>>>(nbf, rm, rl, 0);
  attn_stats_kernel<<<128, 256, 0, stream>>>(nbf, cm, cl, 1);

  // 4) attention apply -> attention_input = nodes - cross
  attn_apply_kernel<<<1024, 256, 0, stream>>>(nbf, nodes, rm, rl, attin, 0);
  attn_apply_kernel<<<1024, 256, 0, stream>>>(nbf, nodes, cm, cl, attin, 1);

  // 5) node MLP with residual -> d_out
  node_mlp_kernel<<<N_NODES / NPB, 512, 0, stream>>>(
      nodes, nbf, agg, attin, wn1b, bn1, wn2b, bn2, out);
}
